// AngularAttention_10685878633159
// MI455X (gfx1250) — matrix operand order repaired and checked by reading, not compiled, not run on hardware
//
#include <hip/hip_runtime.h>

typedef __attribute__((ext_vector_type(16))) _Float16 v16h;
typedef __attribute__((ext_vector_type(8)))  _Float16 v8h;
typedef __attribute__((ext_vector_type(8)))  float    v8f;

// ---------------------------------------------------------------------------
// WMMA helpers (CDNA5 16x16x32 f16 -> f32)
// ---------------------------------------------------------------------------
__device__ __forceinline__ v8f wmma16(v16h a, v16h b, v8f c) {
  return __builtin_amdgcn_wmma_f32_16x16x32_f16(
      /*neg_a=*/false, a, /*neg_b=*/false, b,
      /*c_mod=*/(short)0, c, /*reuse_a=*/false, /*reuse_b=*/false);
}

// A-matrix 16x32 f16 fragment from a row-major [rows][stride] f16 buffer.
// Per ISA 7.12.2: lane l holds row m0+(l&15); elements 0..7 -> K = k0+8h+e,
// elements 8..15 -> K = k0+16+8h+(e-8)   (h = l>>4).
__device__ __forceinline__ v16h load_a_frag(const _Float16* base, int stride,
                                            int m0, int k0, int lane) {
  const int m = m0 + (lane & 15);
  const int h = lane >> 4;
  v8h lo = *(const v8h*)(base + (size_t)m * stride + k0 + 8 * h);
  v8h hi = *(const v8h*)(base + (size_t)m * stride + k0 + 16 + 8 * h);
  v16h a;
#pragma unroll
  for (int i = 0; i < 8; ++i) { a[i] = lo[i]; a[i + 8] = hi[i]; }
  return a;
}

// B-matrix 32x16 f16 fragment. Source is "N-major": base[n][k] with K
// contiguous (i.e. the transposed weight / k row-major / vT). Per the sparse-B
// layout: lane l holds column n0+(l&15); elements 0..7 -> K = k0+8h+e,
// elements 8..15 -> K = k0+16+8h+(e-8)   (h = l>>4).
__device__ __forceinline__ v16h load_b_frag(const _Float16* base, int stride,
                                            int n0, int k0, int lane) {
  const int n = n0 + (lane & 15);
  const int h = lane >> 4;
  v8h lo = *(const v8h*)(base + (size_t)n * stride + k0 + 8 * h);
  v8h hi = *(const v8h*)(base + (size_t)n * stride + k0 + 16 + 8 * h);
  v16h b;
#pragma unroll
  for (int i = 0; i < 8; ++i) { b[i] = lo[i]; b[i + 8] = hi[i]; }
  return b;
}

// ---------------------------------------------------------------------------
// Conversion kernels
// ---------------------------------------------------------------------------
__global__ void cvt_f32_to_f16(const float* __restrict__ src,
                               _Float16* __restrict__ dst, int n) {
  int i = blockIdx.x * blockDim.x + threadIdx.x;
  if (i < n) dst[i] = (_Float16)src[i];
}

// w [K][N] (row-major) -> wT [N][K] f16
__global__ void cvt_wT(const float* __restrict__ w, _Float16* __restrict__ wT,
                       int K, int N) {
  int i = blockIdx.x * blockDim.x + threadIdx.x;
  if (i < K * N) {
    int n = i / K, k = i % K;
    wT[i] = (_Float16)w[(size_t)k * N + n];
  }
}

// ---------------------------------------------------------------------------
// GEMM: C[M][N] = A[M][K](f16) * BT[N][K](f16)^T + bias.  Block = 128x64,
// 8 waves of 32x32 each.  Output either f16 (intermediate) or f32 (final).
// ---------------------------------------------------------------------------
__launch_bounds__(256)
__global__ void gemm_wmma(const _Float16* __restrict__ A,
                          const _Float16* __restrict__ BT,
                          const float* __restrict__ bias,
                          float* __restrict__ Cf32,
                          _Float16* __restrict__ Cf16, int N, int K) {
  const int lane = threadIdx.x & 31;
  const int wave = threadIdx.x >> 5;
  const int m0 = blockIdx.x * 128 + (wave & 3) * 32;
  const int n0 = blockIdx.y * 64 + (wave >> 2) * 32;

  v8f acc00 = {}, acc01 = {}, acc10 = {}, acc11 = {};
  for (int k0 = 0; k0 < K; k0 += 32) {
    v16h a0 = load_a_frag(A, K, m0, k0, lane);
    v16h a1 = load_a_frag(A, K, m0 + 16, k0, lane);
    v16h b0 = load_b_frag(BT, K, n0, k0, lane);
    v16h b1 = load_b_frag(BT, K, n0 + 16, k0, lane);
    acc00 = wmma16(a0, b0, acc00);
    acc01 = wmma16(a0, b1, acc01);
    acc10 = wmma16(a1, b0, acc10);
    acc11 = wmma16(a1, b1, acc11);
  }
  const int h = lane >> 4, nc = lane & 15;
  const float bv0 = bias[n0 + nc];
  const float bv1 = bias[n0 + 16 + nc];
#pragma unroll
  for (int r = 0; r < 8; ++r) {
    const int mA = m0 + r + 8 * h;
    const int mB = m0 + 16 + r + 8 * h;
    const float v00 = acc00[r] + bv0, v01 = acc01[r] + bv1;
    const float v10 = acc10[r] + bv0, v11 = acc11[r] + bv1;
    if (Cf16) {
      Cf16[(size_t)mA * N + n0 + nc]      = (_Float16)v00;
      Cf16[(size_t)mA * N + n0 + 16 + nc] = (_Float16)v01;
      Cf16[(size_t)mB * N + n0 + nc]      = (_Float16)v10;
      Cf16[(size_t)mB * N + n0 + 16 + nc] = (_Float16)v11;
    } else {
      Cf32[(size_t)mA * N + n0 + nc]      = v00;
      Cf32[(size_t)mA * N + n0 + 16 + nc] = v01;
      Cf32[(size_t)mB * N + n0 + nc]      = v10;
      Cf32[(size_t)mB * N + n0 + 16 + nc] = v11;
    }
  }
}

// ---------------------------------------------------------------------------
// Windowed cosine attention, one block per (window, head).
// L = 196 tokens padded to 224 (14 tiles).  13 waves; wave w owns query rows
// [16w, 16w+16).  Flash-style online softmax over 7 K-blocks of 32 columns.
// ---------------------------------------------------------------------------
#define L_TOK 196
#define LP    224
#define HD    32

__launch_bounds__(416)
__global__ void attn_kernel(const _Float16* __restrict__ qkv,
                            _Float16* __restrict__ attnout) {
  __shared__ _Float16 sQ[LP * HD];        // [token][dim]
  __shared__ _Float16 sK[LP * HD];        // [token][dim]
  __shared__ _Float16 sVT[HD * LP];       // [dim][token]
  __shared__ _Float16 sP[13 * 16 * HD];   // per-wave P scratch [16][32]

  const int tid  = threadIdx.x;
  const int lane = tid & 31;
  const int wave = tid >> 5;              // 0..12
  const int head = blockIdx.x & 7;
  const int win  = blockIdx.x >> 3;
  const int b  = win >> 4;
  const int wy = (win >> 2) & 3;
  const int wx = win & 3;

  // ---- stage q,k,v into LDS (zero-pad tokens >= 196) ----
  for (int u = tid; u < LP * 4; u += 416) {
    const int t  = u >> 2;
    const int dc = (u & 3) * 8;
    v8h q8, k8, v8;
    if (t < L_TOK) {
      const int ty = t / 14, tx = t % 14;
      const int gh = wy * 14 + ty, gw = wx * 14 + tx;
      const size_t base =
          ((size_t)(b * 56 + gh) * 56 + gw) * 768 + head * 32 + dc;
      q8 = *(const v8h*)(qkv + base);
      k8 = *(const v8h*)(qkv + base + 256);
      v8 = *(const v8h*)(qkv + base + 512);
    } else {
#pragma unroll
      for (int i = 0; i < 8; ++i) { q8[i] = (_Float16)0; k8[i] = (_Float16)0; v8[i] = (_Float16)0; }
    }
    *(v8h*)(&sQ[t * HD + dc]) = q8;
    *(v8h*)(&sK[t * HD + dc]) = k8;
#pragma unroll
    for (int i = 0; i < 8; ++i) sVT[(dc + i) * LP + t] = v8[i];
  }
  __syncthreads();

  // ---- L2-normalize q and k rows (cosine attention) ----
  for (int r = tid; r < 2 * LP; r += 416) {
    _Float16* row = (r < LP) ? &sQ[r * HD] : &sK[(r - LP) * HD];
    float ss = 0.f;
#pragma unroll
    for (int i = 0; i < HD; ++i) { float x = (float)row[i]; ss += x * x; }
    const float sc = 1.0f / fmaxf(sqrtf(ss), 1e-12f);
#pragma unroll
    for (int i = 0; i < HD; ++i) row[i] = (_Float16)((float)row[i] * sc);
  }
  __syncthreads();

  // ---- per-wave flash attention over its 16 query rows ----
  const int m0 = wave * 16;
  const v16h aQ = load_a_frag(sQ, HD, m0, 0, lane);   // hd==32 -> single A frag

  v8f acc0 = {}, acc1 = {};                           // O tile, dims 0..15 / 16..31
  float rm[8], rs[8];
#pragma unroll
  for (int r = 0; r < 8; ++r) { rm[r] = -3e38f; rs[r] = 0.f; }

  _Float16* pbuf = &sP[wave * 16 * HD];
  const int ncol = lane & 15;
  const int hh   = lane >> 4;
  const float scale = 10.0f;                          // 1 / TEMP

  for (int kb = 0; kb < 7; ++kb) {
    const int c0 = kb * 32;
    const v16h bK0 = load_b_frag(sK, HD, c0,      0, lane);
    const v16h bK1 = load_b_frag(sK, HD, c0 + 16, 0, lane);
    v8f z0 = {}, z1 = {};
    v8f s0 = wmma16(aQ, bK0, z0);
    v8f s1 = wmma16(aQ, bK1, z1);
    const bool valid0 = (c0 + ncol)      < L_TOK;
    const bool valid1 = (c0 + 16 + ncol) < L_TOK;

#pragma unroll
    for (int r = 0; r < 8; ++r) {
      float x0 = valid0 ? s0[r] * scale : -3e38f;
      float x1 = valid1 ? s1[r] * scale : -3e38f;
      // row max over the 16 lanes of this half (rows r / r+8 live in halves)
      float mx = fmaxf(x0, x1);
#pragma unroll
      for (int d = 8; d >= 1; d >>= 1) mx = fmaxf(mx, __shfl_xor(mx, d, 32));
      const float mnew = fmaxf(rm[r], mx);
      const float corr = __expf(rm[r] - mnew);
      const float p0 = __expf(x0 - mnew);
      const float p1 = __expf(x1 - mnew);
      float ps = p0 + p1;
#pragma unroll
      for (int d = 8; d >= 1; d >>= 1) ps += __shfl_xor(ps, d, 32);
      rs[r] = rs[r] * corr + ps;
      rm[r] = mnew;
      acc0[r] *= corr;
      acc1[r] *= corr;
      // stash P tile (C layout -> row-major LDS) for the A-layout reload
      const int prow = r + 8 * hh;
      pbuf[prow * HD + ncol]      = (_Float16)p0;
      pbuf[prow * HD + 16 + ncol] = (_Float16)p1;
    }
    asm volatile("" ::: "memory");  // order LDS P stores before reload

    const v16h aP  = load_a_frag(pbuf, HD, 0, 0, lane);
    const v16h bV0 = load_b_frag(sVT, LP, 0,  c0, lane);
    const v16h bV1 = load_b_frag(sVT, LP, 16, c0, lane);
    acc0 = wmma16(aP, bV0, acc0);
    acc1 = wmma16(aP, bV1, acc1);
  }

  // ---- epilogue: divide by row sums, scatter into [B,H,W,E] f16 ----
#pragma unroll
  for (int r = 0; r < 8; ++r) {
    const int m = m0 + r + 8 * hh;
    if (m >= L_TOK) continue;
    const float inv = 1.0f / rs[r];
    const int ty = m / 14, tx = m % 14;
    const int gh = wy * 14 + ty, gw = wx * 14 + tx;
    const size_t obase = ((size_t)(b * 56 + gh) * 56 + gw) * 256 + head * 32;
    attnout[obase + ncol]      = (_Float16)(acc0[r] * inv);
    attnout[obase + 16 + ncol] = (_Float16)(acc1[r] * inv);
  }
}

// ---------------------------------------------------------------------------
// Launch
// ---------------------------------------------------------------------------
extern "C" void kernel_launch(void* const* d_in, const int* in_sizes, int n_in,
                              void* d_out, int out_size, void* d_ws, size_t ws_size,
                              hipStream_t stream) {
  const float* x      = (const float*)d_in[0];  // [16,56,56,256]
  const float* w_qkv  = (const float*)d_in[1];  // [256,768]
  const float* b_qkv  = (const float*)d_in[2];  // [768]
  const float* w_proj = (const float*)d_in[3];  // [256,256]
  const float* b_proj = (const float*)d_in[4];  // [256]
  float* out = (float*)d_out;                   // [16,56,56,256] f32

  const int M = 16 * 56 * 56;                   // 50176 tokens
  char* ws = (char*)d_ws;
  _Float16* xh     = (_Float16*)(ws);                           // 25,690,112 B
  _Float16* wTqkv  = (_Float16*)(ws + 25690112);                //    393,216 B
  _Float16* wTproj = (_Float16*)(ws + 26083328);                //    131,072 B
  _Float16* qkvh   = (_Float16*)(ws + 26214400);                // 77,070,336 B
  _Float16* aouth  = (_Float16*)(ws + 103284736);               // 25,690,112 B

  const int nx = M * 256;
  cvt_f32_to_f16<<<(nx + 255) / 256, 256, 0, stream>>>(x, xh, nx);
  cvt_wT<<<(256 * 768 + 255) / 256, 256, 0, stream>>>(w_qkv, wTqkv, 256, 768);
  cvt_wT<<<(256 * 256 + 255) / 256, 256, 0, stream>>>(w_proj, wTproj, 256, 256);

  // qkv = x @ w_qkv + b_qkv   (M x 768, K=256) -> f16
  gemm_wmma<<<dim3(M / 128, 768 / 64), 256, 0, stream>>>(
      xh, wTqkv, b_qkv, nullptr, qkvh, 768, 256);

  // windowed cosine attention: 256 windows x 8 heads
  attn_kernel<<<2048, 416, 0, stream>>>(qkvh, aouth);

  // out = attn_out @ w_proj + b_proj  (M x 256, K=256) -> f32
  gemm_wmma<<<dim3(M / 128, 256 / 64), 256, 0, stream>>>(
      aouth, wTproj, b_proj, out, nullptr, 256, 256);
}
